// ASTGCN_submodule_63333587747454
// MI455X (gfx1250) — matrix operand order repaired and checked by reading, not compiled, not run on hardware
//
#include <hip/hip_runtime.h>
#include <hip/hip_bf16.h>

typedef __attribute__((ext_vector_type(16))) _Float16 v16h;
typedef __attribute__((ext_vector_type(8)))  float    v8f;
typedef __attribute__((ext_vector_type(4)))  int      i4v;
typedef __attribute__((ext_vector_type(4)))  float    f4v;

#define NB      4
#define NN      1024
#define NT      4
#define ALPHA_  0.2f
#define NEGV    (-9.0e15f)

// ---- workspace layout (byte offsets, 256-aligned) ----
#define WS_CONST 0u          // 392 f32: c1[4], c2[4], M[128], biasSum[256]
#define WS_WCOMB 2048u       // 256*96 f16  (per-output-column contiguous K)
#define WS_F1    51200u      // 16384 f32
#define WS_F2    116736u     // 16384 f32
#define WS_WH2   182272u     // 16*32*1024 f16, column-major per (b,t)
#define WS_SG    1230848u    // 16*1024*32 f16, row-major
#define WS_H     2279424u    // 4096*64 f16
#define WS_C     2803712u    // 4096*64 f32
#define WS_LT    3852288u    // 16*1024*64 f32

// ---------------------------------------------------------------------------
// Phase A: collapse parameters.
// ---------------------------------------------------------------------------
__global__ void k_prep(const float* __restrict__ headsW, const float* __restrict__ headsA,
                       const float* __restrict__ outW,
                       const float* __restrict__ Wih, const float* __restrict__ Whh,
                       const float* __restrict__ bih, const float* __restrict__ bhh,
                       float* __restrict__ cst, _Float16* __restrict__ wcomb) {
    const int tid = threadIdx.x;               // 256 threads
    const int o = tid;
    for (int k = 0; k < 32; ++k) wcomb[o*96 + k]      = (_Float16)Wih[o*32 + k];
    for (int k = 0; k < 64; ++k) wcomb[o*96 + 32 + k] = (_Float16)Whh[o*64 + k];
    cst[136 + o] = bih[o] + bhh[o];
    if (tid < 128) {
        int h = tid >> 5, o2 = tid & 31;
        float s = 0.f;
        for (int j = 0; j < 16; ++j) s += headsW[h*16 + j] * outW[(h*16 + j)*32 + o2];
        cst[8 + h*32 + o2] = s;
    }
    if (tid < 8) {
        int h = tid & 3, hi = tid >> 2;
        float s = 0.f;
        for (int j = 0; j < 16; ++j) s += headsW[h*16 + j] * headsA[h*32 + hi*16 + j];
        cst[hi*4 + h] = s;
    }
}

// ---------------------------------------------------------------------------
// Phase B: 4 head-GATs fused (online softmax, scalar values), then
// Wh2 row = S @ M, plus layer-2 attention node scalars f1/f2.
// ---------------------------------------------------------------------------
__global__ void k_heads(const float* __restrict__ x, const int* __restrict__ adj,
                        const float* __restrict__ outA, const float* __restrict__ cst,
                        float* __restrict__ f1b, float* __restrict__ f2b,
                        _Float16* __restrict__ wh2cm) {
    const int lane = threadIdx.x & 31;
    const int w    = threadIdx.x >> 5;
    const int r  = blockIdx.x * 4 + w;          // 0..16383  = ((b*T+t)*N + n)
    const int n  = r & (NN - 1);
    const int bt = r >> 10;
    const int b  = bt >> 2, t = bt & 3;
    const float* xb = x + (size_t)b * NN * 12 + t;
    const float xn  = xb[(size_t)n * 12];
    float c1[4], c2[4];
#pragma unroll
    for (int h = 0; h < 4; ++h) { c1[h] = cst[h]; c2[h] = cst[4 + h]; }
    const int* arow = adj + (size_t)n * NN;

    float mx[4], l[4], ac[4];
#pragma unroll
    for (int h = 0; h < 4; ++h) { mx[h] = NEGV; l[h] = 0.f; ac[h] = 0.f; }
    for (int i = 0; i < 32; ++i) {
        int m = lane + i * 32;
        int aij = arow[m];
        float xm = xb[(size_t)m * 12];
#pragma unroll
        for (int h = 0; h < 4; ++h) {
            float s = c1[h] * xn + c2[h] * xm;
            s = s > 0.f ? s : ALPHA_ * s;
            s = aij > 0 ? s : NEGV;
            float nm = fmaxf(mx[h], s);
            float sc = __expf(mx[h] - nm);
            float p  = __expf(s - nm);
            l[h]  = l[h] * sc + p;
            ac[h] = ac[h] * sc + p * xm;
            mx[h] = nm;
        }
    }
#pragma unroll
    for (int off = 16; off > 0; off >>= 1) {
#pragma unroll
        for (int h = 0; h < 4; ++h) {
            float om = __shfl_xor(mx[h], off, 32);
            float ol = __shfl_xor(l[h],  off, 32);
            float oa = __shfl_xor(ac[h], off, 32);
            float nm = fmaxf(mx[h], om);
            float e1 = __expf(mx[h] - nm), e2 = __expf(om - nm);
            l[h]  = l[h]  * e1 + ol * e2;
            ac[h] = ac[h] * e1 + oa * e2;
            mx[h] = nm;
        }
    }
    float S[4];
#pragma unroll
    for (int h = 0; h < 4; ++h) S[h] = ac[h] / l[h];

    const float* Mc = cst + 8;
    float wv = S[0]*Mc[lane] + S[1]*Mc[32+lane] + S[2]*Mc[64+lane] + S[3]*Mc[96+lane];
    wh2cm[((size_t)bt * 32 + lane) * NN + n] = (_Float16)wv;
    float p1 = wv * outA[lane];
    float p2 = wv * outA[32 + lane];
#pragma unroll
    for (int off = 16; off > 0; off >>= 1) {
        p1 += __shfl_xor(p1, off, 32);
        p2 += __shfl_xor(p2, off, 32);
    }
    if (lane == 0) { f1b[bt * NN + n] = p1; f2b[bt * NN + n] = p2; }
}

// ---------------------------------------------------------------------------
// Phase C: out-GAT, flash-style with async LDS staging.
// The 64KB Wh2 panel + 4KB f2 row for this (b,t) are shared by all 4 waves,
// so stage them once per block with GLOBAL_LOAD_ASYNC_TO_LDS_B128 (ASYNCcnt),
// then WMMA B operands come from LDS (ds_load_b128).
// ---------------------------------------------------------------------------
__global__ void k_gat2(const int* __restrict__ adj,
                       const float* __restrict__ f1b, const float* __restrict__ f2b,
                       const _Float16* __restrict__ wh2cm,
                       _Float16* __restrict__ sg) {
    extern __shared__ __align__(128) char dynsm[];   // 64KB panel + 4KB f2
    _Float16* panel = (_Float16*)dynsm;              // [32][1024] f16
    float*    f2s   = (float*)(dynsm + 65536);       // [1024] f32
    __shared__ float smax[4][16];
    __shared__ float sinv[4][16];

    const int lane  = threadIdx.x & 31;
    const int w     = threadIdx.x >> 5;
    const int khalf = lane >> 4;
    const int lrow  = lane & 15;
    const int bt = blockIdx.y;
    const int rowTile = blockIdx.x * 4 + w;       // 0..63
    const float* f1r = f1b + bt * NN;

    // ---- async stage: wh2 panel (4096 x 16B) + f2 row (256 x 16B) ----
    {
        const char* gpanel = (const char*)(wh2cm + (size_t)bt * 32 * NN);
        const char* gf2    = (const char*)(f2b + (size_t)bt * NN);
        uint32_t lpanel = (uint32_t)(uintptr_t)panel;
        uint32_t lf2    = (uint32_t)(uintptr_t)f2s;
#pragma unroll 4
        for (int i = 0; i < 32; ++i) {
            int c = i * 128 + threadIdx.x;
            uint32_t laddr = lpanel + c * 16;
            const char* g = gpanel + (size_t)c * 16;
            asm volatile("global_load_async_to_lds_b128 %0, %1, off"
                         :: "v"(laddr), "v"(g) : "memory");
        }
        {
            int c = threadIdx.x;
            asm volatile("global_load_async_to_lds_b128 %0, %1, off"
                         :: "v"(lf2 + c * 16), "v"(gf2 + (size_t)c * 16) : "memory");
            asm volatile("global_load_async_to_lds_b128 %0, %1, off"
                         :: "v"(lf2 + (c + 128) * 16), "v"(gf2 + (size_t)(c + 128) * 16) : "memory");
        }
        asm volatile("s_wait_asynccnt 0x0" ::: "memory");
        __syncthreads();
    }

    // ---- pass 1: per-row softmax stats ----
    for (int r16 = 0; r16 < 16; ++r16) {
        int nrow = rowTile * 16 + r16;
        const int* arow = adj + (size_t)nrow * NN;
        float f1v = f1r[nrow];
        float mx = NEGV, sum = 0.f;
        for (int i = 0; i < 32; ++i) {
            int m = lane + i * 32;
            float s = f1v + f2s[m];
            s = s > 0.f ? s : ALPHA_ * s;
            s = arow[m] > 0 ? s : NEGV;
            float nm = fmaxf(mx, s);
            sum = sum * __expf(mx - nm) + __expf(s - nm);
            mx = nm;
        }
#pragma unroll
        for (int off = 16; off > 0; off >>= 1) {
            float om = __shfl_xor(mx, off, 32);
            float os = __shfl_xor(sum, off, 32);
            float nm = fmaxf(mx, om);
            sum = sum * __expf(mx - nm) + os * __expf(om - nm);
            mx = nm;
        }
        if (lane == r16) { smax[w][r16] = mx; sinv[w][r16] = 1.f / sum; }
    }
    __syncthreads();

    const float rmaxL = smax[w][lrow];
    const int   nA    = rowTile * 16 + lrow;
    const float f1v   = f1r[nA];
    const int*  arowA = adj + (size_t)nA * NN;

    v8f acc0 = {}; v8f acc1 = {};
    for (int kt = 0; kt < 32; ++kt) {
        int m0 = kt * 32;
        int mA = m0 + khalf * 16;
        int mB = m0 + khalf * 16 + 8;
        // vectorized adj / f2 fetch for the two 8-wide runs
        i4v a0 = ((const i4v*)(arowA + mA))[0];
        i4v a1 = ((const i4v*)(arowA + mA))[1];
        i4v a2 = ((const i4v*)(arowA + mB))[0];
        i4v a3 = ((const i4v*)(arowA + mB))[1];
        f4v q0 = ((const f4v*)(f2s + mA))[0];
        f4v q1 = ((const f4v*)(f2s + mA))[1];
        f4v q2 = ((const f4v*)(f2s + mB))[0];
        f4v q3 = ((const f4v*)(f2s + mB))[1];
        v16h a;
#pragma unroll
        for (int e = 0; e < 8; ++e) {
            int   aij = (e < 4) ? a0[e] : a1[e - 4];
            float f2v = (e < 4) ? q0[e] : q1[e - 4];
            float s = f1v + f2v;
            s = s > 0.f ? s : ALPHA_ * s;
            s = aij > 0 ? s : NEGV;
            a[e] = (_Float16)__expf(s - rmaxL);
        }
#pragma unroll
        for (int e = 0; e < 8; ++e) {
            int   aij = (e < 4) ? a2[e] : a3[e - 4];
            float f2v = (e < 4) ? q2[e] : q3[e - 4];
            float s = f1v + f2v;
            s = s > 0.f ? s : ALPHA_ * s;
            s = aij > 0 ? s : NEGV;
            a[8 + e] = (_Float16)__expf(s - rmaxL);
        }
        // B tiles from LDS: col = lrow, K = khalf*16 + e (contiguous m)
        const _Float16* bp0 = panel + (size_t)lrow * NN        + m0 + khalf * 16;
        const _Float16* bp1 = panel + (size_t)(16 + lrow) * NN + m0 + khalf * 16;
        v16h b0 = *(const v16h*)bp0;
        v16h b1 = *(const v16h*)bp1;
        acc0 = __builtin_amdgcn_wmma_f32_16x16x32_f16(false, a, false, b0, (short)0, acc0, false, false);
        acc1 = __builtin_amdgcn_wmma_f32_16x16x32_f16(false, a, false, b1, (short)0, acc1, false, false);
    }
#pragma unroll
    for (int rr = 0; rr < 8; ++rr) {
        int Mrow = rr + 8 * khalf;
        float inv = sinv[w][Mrow];
        int n = rowTile * 16 + Mrow;
        sg[((size_t)bt * NN + n) * 32 + lrow]      = (_Float16)(acc0[rr] * inv);
        sg[((size_t)bt * NN + n) * 32 + 16 + lrow] = (_Float16)(acc1[rr] * inv);
    }
}

// ---------------------------------------------------------------------------
// Phase D: one LSTM step. gates(16x256) = [X_t | h](16x96) @ Wcomb^T via WMMA.
// ---------------------------------------------------------------------------
__global__ void k_lstm(const _Float16* __restrict__ sg, const _Float16* __restrict__ wcomb,
                       const float* __restrict__ cst,
                       _Float16* __restrict__ hbuf, float* __restrict__ cbuf,
                       float* __restrict__ ltbuf, int t) {
    __shared__ float gbuf[16 * 256];
    const int lane  = threadIdx.x & 31;
    const int w     = threadIdx.x >> 5;
    const int khalf = lane >> 4;
    const int lrow  = lane & 15;
    const int rt = blockIdx.x;                  // 0..255 (row tile)
    const int rr = rt * 16 + lrow;              // A row = b*N + n
    const int b  = rr >> 10, n = rr & 1023;
    const _Float16* sgrow = sg + ((size_t)(b * NT + t) * NN + n) * 32;
    const _Float16* hrow  = hbuf + (size_t)rr * 64;

    v16h A[3];
#pragma unroll
    for (int kt2 = 0; kt2 < 3; ++kt2) {
#pragma unroll
        for (int e = 0; e < 16; ++e) {
            int klocal = khalf * 16 + e;
            int k = kt2 * 32 + klocal;
            _Float16 v;
            if (k < 32)      v = sgrow[k];
            else if (t == 0) v = (_Float16)0.f;
            else             v = hrow[k - 32];
            A[kt2][e] = v;
        }
    }
    const float* biasS = cst + 136;
    v8f acc[4] = {};
#pragma unroll
    for (int ot = 0; ot < 4; ++ot) {
        int col = w * 64 + ot * 16 + lrow;
        const _Float16* wp = wcomb + (size_t)col * 96 + khalf * 16;
#pragma unroll
        for (int kt2 = 0; kt2 < 3; ++kt2) {
            v16h bb = *(const v16h*)(wp + kt2 * 32);
            acc[ot] = __builtin_amdgcn_wmma_f32_16x16x32_f16(false, A[kt2], false, bb, (short)0, acc[ot], false, false);
        }
        float bsv = biasS[col];
#pragma unroll
        for (int r8 = 0; r8 < 8; ++r8) {
            int Mrow = r8 + 8 * khalf;
            gbuf[Mrow * 256 + col] = acc[ot][r8] + bsv;
        }
    }
    __syncthreads();
#pragma unroll
    for (int q = 0; q < 8; ++q) {
        int idx = threadIdx.x + q * 128;        // 0..1023 = 16 rows x 64 hidden
        int row = idx >> 6, j = idx & 63;
        int rr2 = rt * 16 + row;
        float gi = gbuf[row * 256 + j];
        float gf = gbuf[row * 256 + 64 + j];
        float gg = gbuf[row * 256 + 128 + j];
        float go = gbuf[row * 256 + 192 + j];
        float cold = (t == 0) ? 0.f : cbuf[(size_t)rr2 * 64 + j];
        float si = 1.f / (1.f + __expf(-gi));
        float sf = 1.f / (1.f + __expf(-gf));
        float so = 1.f / (1.f + __expf(-go));
        float cnew = sf * cold + si * tanhf(gg);
        float hv   = so * tanhf(cnew);
        cbuf[(size_t)rr2 * 64 + j] = cnew;
        hbuf[(size_t)rr2 * 64 + j] = (_Float16)hv;
        int b2 = rr2 >> 10, n2 = rr2 & 1023;
        ltbuf[((size_t)(b2 * NT + t) * NN + n2) * 64 + j] = hv;
    }
}

// ---------------------------------------------------------------------------
// Phase E: conv residual + ReLU + LayerNorm(64) + time-mix + finb.
// ---------------------------------------------------------------------------
__global__ void k_final(const float* __restrict__ x, const float* __restrict__ ltbuf,
                        const float* __restrict__ convW, const float* __restrict__ convb,
                        const float* __restrict__ lng, const float* __restrict__ lnb,
                        const float* __restrict__ finW, const float* __restrict__ finb,
                        float* __restrict__ out) {
    __shared__ float sh1[2], sh2[2];
    const int f = threadIdx.x;                  // 0..63
    const int lane = f & 31, wid = f >> 5;
    const int bn = blockIdx.x;
    const int b = bn >> 10, n = bn & 1023;
    const float cw = convW[f], cb = convb[f];
    const float g  = lng[f],  be = lnb[f];
    float op[4] = {0.f, 0.f, 0.f, 0.f};
    for (int t = 0; t < 4; ++t) {
        float xv = x[(size_t)(b * NN + n) * 12 + t];
        float v = xv * cw + cb + ltbuf[((size_t)(b * NT + t) * NN + n) * 64 + f];
        v = fmaxf(v, 0.f);
        float s1 = v, s2 = v * v;
#pragma unroll
        for (int off = 16; off > 0; off >>= 1) {
            s1 += __shfl_xor(s1, off, 32);
            s2 += __shfl_xor(s2, off, 32);
        }
        if (lane == 0) { sh1[wid] = s1; sh2[wid] = s2; }
        __syncthreads();
        float S1 = sh1[0] + sh1[1];
        float S2 = sh2[0] + sh2[1];
        __syncthreads();
        float mu  = S1 * (1.f / 64.f);
        float var = S2 * (1.f / 64.f) - mu * mu;
        float y = (v - mu) * rsqrtf(var + 1e-5f) * g + be;
#pragma unroll
        for (int p = 0; p < 4; ++p) op[p] += y * finW[p * 4 + t];
    }
#pragma unroll
    for (int p = 0; p < 4; ++p)
        out[((size_t)(b * NN + n) * 64 + f) * 4 + p] = op[p] + finb[p];
}

// ---------------------------------------------------------------------------
extern "C" void kernel_launch(void* const* d_in, const int* in_sizes, int n_in,
                              void* d_out, int out_size, void* d_ws, size_t ws_size,
                              hipStream_t stream) {
    (void)in_sizes; (void)n_in; (void)out_size; (void)ws_size;
    const float* x      = (const float*)d_in[0];
    const int*   adj    = (const int*)  d_in[1];
    const float* headsW = (const float*)d_in[2];
    const float* headsA = (const float*)d_in[3];
    const float* outW   = (const float*)d_in[4];
    const float* outA   = (const float*)d_in[5];
    const float* Wih    = (const float*)d_in[6];
    const float* Whh    = (const float*)d_in[7];
    const float* bih    = (const float*)d_in[8];
    const float* bhh    = (const float*)d_in[9];
    const float* convW  = (const float*)d_in[10];
    const float* convb  = (const float*)d_in[11];
    const float* lng    = (const float*)d_in[12];
    const float* lnb    = (const float*)d_in[13];
    const float* finW   = (const float*)d_in[14];
    const float* finb   = (const float*)d_in[15];
    float* out = (float*)d_out;

    char* wsb = (char*)d_ws;
    float*    cst   = (float*)   (wsb + WS_CONST);
    _Float16* wcomb = (_Float16*)(wsb + WS_WCOMB);
    float*    f1b   = (float*)   (wsb + WS_F1);
    float*    f2b   = (float*)   (wsb + WS_F2);
    _Float16* wh2   = (_Float16*)(wsb + WS_WH2);
    _Float16* sgf   = (_Float16*)(wsb + WS_SG);
    _Float16* hbuf  = (_Float16*)(wsb + WS_H);
    float*    cbuf  = (float*)   (wsb + WS_C);
    float*    ltbuf = (float*)   (wsb + WS_LT);

    k_prep <<<1, 256, 0, stream>>>(headsW, headsA, outW, Wih, Whh, bih, bhh, cst, wcomb);
    k_heads<<<4096, 128, 0, stream>>>(x, adj, outA, cst, f1b, f2b, wh2);
    k_gat2 <<<dim3(16, 16), 128, 69632, stream>>>(adj, f1b, f2b, wh2, sgf);
    for (int t = 0; t < 4; ++t)
        k_lstm<<<256, 128, 0, stream>>>(sgf, wcomb, cst, hbuf, cbuf, ltbuf, t);
    k_final<<<4096, 64, 0, stream>>>(x, ltbuf, convW, convb, lng, lnb, finW, finb, out);
}
